// WeaveLayer_33904471834980
// MI455X (gfx1250) — matrix operand order repaired and checked by reading, not compiled, not run on hardware
//
#include <hip/hip_runtime.h>
#include <hip/hip_bf16.h>

typedef __bf16 v16bf __attribute__((ext_vector_type(16)));
typedef __bf16 v8bf  __attribute__((ext_vector_type(8)));
typedef float  v8f   __attribute__((ext_vector_type(8)));

#define B_SZ   64
#define M_SZ   128
#define F_ATOM 75
#define F_PAIR 14
#define HID    50
#define OUT_A  50
#define OUT_P  50

// WMMA-padded dims
#define KAP  96    // 75 -> 96  (3 x K=32)
#define KPP  32    // 14 -> 32  (1 x K=32)
#define KP   128   // hidden: AP block [0,64) ++ PP block [64,128)
#define NPAD 64    // 50 -> 64  (4 x N=16)

// packed-weight (bf16) workspace layout, TRANSPOSED [n][k], element offsets
#define WT_WAP 0
#define WT_WPP (NPAD * KAP)
#define WT_WP  (NPAD * KAP + NPAD * KPP)
#define WT_TOT (NPAD * KAP + NPAD * KPP + NPAD * KP)   // 16384 bf16 = 32 KB

// --------------------------------------------------------------------------
// Kernel 0: fold W_AP (concat trick), zero-pad, TRANSPOSE to [n][k] bf16.
// W_P rows are remapped to the padded hidden layout: k'<64 -> AP row k'
// (zero for k'>=50), k'>=64 -> PP row (k'-64) (zero for k'-64>=50).
// --------------------------------------------------------------------------
__global__ __launch_bounds__(256) void pack_weights(
    const float* __restrict__ W_AP, const float* __restrict__ W_PP,
    const float* __restrict__ W_P, __bf16* __restrict__ ws)
{
    int idx = blockIdx.x * blockDim.x + threadIdx.x;
    if (idx >= WT_TOT) return;
    float v = 0.f;
    if (idx < WT_WPP) {
        int h = idx / KAP, k = idx % KAP;
        if (k < F_ATOM && h < HID)
            v = W_AP[k * HID + h] + W_AP[(F_ATOM + k) * HID + h];
    } else if (idx < WT_WP) {
        int r = idx - WT_WPP;
        int h = r / KPP, k = r % KPP;
        if (k < F_PAIR && h < HID) v = W_PP[k * HID + h];
    } else {
        int r = idx - WT_WP;
        int h = r / KP, k = r % KP;
        if (h < OUT_P) {
            if (k < NPAD) {                      // AP block
                if (k < HID) v = W_P[k * OUT_P + h];
            } else {                             // PP block
                int kk = k - NPAD;
                if (kk < HID) v = W_P[(HID + kk) * OUT_P + h];
            }
        }
    }
    ws[idx] = (__bf16)v;
}

// --------------------------------------------------------------------------
// Kernel 1: atom branch (tiny). One 64-thread block per (b,i).
// --------------------------------------------------------------------------
__global__ __launch_bounds__(64) void atom_kernel(
    const float* __restrict__ atom, const float* __restrict__ pairf,
    const float* __restrict__ atom_mask,
    const float* __restrict__ W_AA, const float* __restrict__ b_AA,
    const float* __restrict__ W_PA, const float* __restrict__ b_PA,
    const float* __restrict__ W_A,  const float* __restrict__ b_A,
    float* __restrict__ Aout)
{
    __shared__ float sS[F_PAIR];
    __shared__ float sH2[2 * HID];
    const int b = blockIdx.x / M_SZ;
    const int i = blockIdx.x % M_SZ;
    const int t = threadIdx.x;

    if (t < F_PAIR) sS[t] = 0.f;
    __syncthreads();

    // sum over j of pair[b,i,j,:]
    {
        const float* src = pairf + ((size_t)(b * M_SZ + i)) * M_SZ * F_PAIR;
        float acc[F_PAIR];
        #pragma unroll
        for (int f = 0; f < F_PAIR; f++) acc[f] = 0.f;
        #pragma unroll
        for (int jj = 0; jj < 2; jj++) {
            const float* row = src + (size_t)(t * 2 + jj) * F_PAIR;
            #pragma unroll
            for (int f = 0; f < F_PAIR; f++) acc[f] += row[f];
        }
        #pragma unroll
        for (int f = 0; f < F_PAIR; f++) atomicAdd(&sS[f], acc[f]);
    }
    __syncthreads();

    if (t < HID) {
        const float* af = atom + (size_t)(b * M_SZ + i) * F_ATOM;
        float aa = b_AA[t];
        for (int f = 0; f < F_ATOM; f++) aa += af[f] * W_AA[f * HID + t];
        float pa = (float)M_SZ * b_PA[t];     // bias summed M times before relu
        #pragma unroll
        for (int f = 0; f < F_PAIR; f++) pa += sS[f] * W_PA[f * HID + t];
        sH2[t]       = fmaxf(aa, 0.f);
        sH2[HID + t] = fmaxf(pa, 0.f);
    }
    __syncthreads();

    if (t < OUT_A) {
        float acc = b_A[t];
        for (int h = 0; h < 2 * HID; h++) acc += sH2[h] * W_A[h * OUT_A + t];
        Aout[(size_t)(b * M_SZ + i) * OUT_A + t] =
            fmaxf(acc, 0.f) * atom_mask[b * M_SZ + i];
    }
}

// combine two contiguous 8-element halves into a 16-element fragment
__device__ __forceinline__ v16bf combine16(v8bf lo, v8bf hi)
{
    return __builtin_shufflevector(lo, hi, 0, 1, 2, 3, 4, 5, 6, 7,
                                   8, 9, 10, 11, 12, 13, 14, 15);
}

// load an A fragment (16x32, bf16) from a row-major bf16 row: two runs
// k = k0+g*8..+7 and k = k0+16+g*8..+7 (documented 16-bit A layout).
__device__ __forceinline__ v16bf load_afrag(const __bf16* row, int k0, int g)
{
    v8bf lo = *(const v8bf*)(row + k0 + g * 8);
    v8bf hi = *(const v8bf*)(row + k0 + 16 + g * 8);
    return combine16(lo, hi);
}

// load a B fragment (32x16, bf16) from a transposed [n][Kdim] LDS matrix:
// lane column c, K base k0; elements e<8 <-> k = k0 + g*8 + e, e>=8 <-> k = k0 + 16 + g*8 + (e-8).
__device__ __forceinline__ v16bf load_bfragT(const __bf16* base, int Kdim,
                                             int c, int k0, int g)
{
    const __bf16* p = base + c * Kdim + k0 + g * 8;
    v8bf lo = *(const v8bf*)(p);
    v8bf hi = *(const v8bf*)(p + 16);
    return combine16(lo, hi);
}

// --------------------------------------------------------------------------
// Kernel 2: pair branch (dominant). One 256-thread block (8 waves) per (b,i);
// each wave owns a 16-row (j) x 64-col tile. bf16 WMMA, f32 accumulate.
// --------------------------------------------------------------------------
__launch_bounds__(256)
__global__ void pair_kernel(
    const float* __restrict__ atom, const float* __restrict__ pairf,
    const float* __restrict__ pair_mask, const __bf16* __restrict__ wpack,
    const float* __restrict__ b_AP, const float* __restrict__ b_PP,
    const float* __restrict__ b_P, float* __restrict__ Pout)
{
    __shared__ __align__(16) __bf16 sW[WT_TOT];        // 32 KB transposed weights
    __shared__ __align__(16) __bf16 sS[M_SZ * KAP];    // 24 KB bf16(a_i + a_j), padded
    __shared__ __align__(16) __bf16 sPB[M_SZ * KPP];   //  8 KB bf16 pair rows, padded
    __shared__ __align__(16) __bf16 sHid[8 * 16 * KP]; // 32 KB hidden tiles

    const int b   = blockIdx.x / M_SZ;
    const int i   = blockIdx.x % M_SZ;
    const int tid = threadIdx.x;

    // stage packed weights (16B vector copies: 2048 uint4)
    {
        const uint4* src = (const uint4*)wpack;
        uint4*       dst = (uint4*)sW;
        for (int idx = tid; idx < WT_TOT / 8; idx += 256) dst[idx] = src[idx];
    }
    // stage s = a_i + a_j as zero-padded bf16 [128][96]
    {
        const float* ai = atom + (size_t)(b * M_SZ + i) * F_ATOM;
        const float* aB = atom + (size_t)b * M_SZ * F_ATOM;
        for (int idx = tid; idx < M_SZ * (KAP / 2); idx += 256) {
            int j  = idx / (KAP / 2);
            int k0 = (idx - j * (KAP / 2)) * 2;
            int k1 = k0 + 1;
            const float* aj = aB + (size_t)j * F_ATOM;
            float x0 = (k0 < F_ATOM) ? (ai[k0] + aj[k0]) : 0.f;
            float x1 = (k1 < F_ATOM) ? (ai[k1] + aj[k1]) : 0.f;
            sS[j * KAP + k0] = (__bf16)x0;
            sS[j * KAP + k1] = (__bf16)x1;
        }
    }
    // stage pair[b,i,:,:] as zero-padded bf16 [128][32]
    {
        const float* src = pairf + ((size_t)(b * M_SZ + i)) * M_SZ * F_PAIR;
        for (int idx = tid; idx < M_SZ * KPP; idx += 256) {
            int j = idx >> 5, f = idx & 31;
            float v = (f < F_PAIR) ? src[j * F_PAIR + f] : 0.f;
            sPB[idx] = (__bf16)v;
        }
    }
    __syncthreads();

    const int wave = tid >> 5;
    const int lane = tid & 31;
    const int g    = lane >> 4;   // lane half (selects K subset / row offset)
    const int r    = lane & 15;   // A-row within tile / B,C column within n-tile
    const int j0   = wave * 16;

    // ---- GEMM 1: AP = s @ Wap, PP = p @ Wpp ----
    v8f cA[4] = {};
    v8f cP[4] = {};
    const __bf16* sRow = &sS[(j0 + r) * KAP];
    const __bf16* pRow = &sPB[(j0 + r) * KPP];

    #pragma unroll
    for (int kc = 0; kc < 3; kc++) {
        v16bf a = load_afrag(sRow, kc * 32, g);
        #pragma unroll
        for (int nt = 0; nt < 4; nt++) {
            v16bf bw = load_bfragT(&sW[WT_WAP], KAP, nt * 16 + r, kc * 32, g);
            cA[nt] = __builtin_amdgcn_wmma_f32_16x16x32_bf16(
                false, a, false, bw, (short)0, cA[nt], false, false);
        }
    }
    {
        v16bf a = load_afrag(pRow, 0, g);
        #pragma unroll
        for (int nt = 0; nt < 4; nt++) {
            v16bf bw = load_bfragT(&sW[WT_WPP], KPP, nt * 16 + r, 0, g);
            cP[nt] = __builtin_amdgcn_wmma_f32_16x16x32_bf16(
                false, a, false, bw, (short)0, cP[nt], false, false);
        }
    }

    // bias + relu, pack hidden tile: AP -> k'[0,64), PP -> k'[64,128).
    // Padded columns (col>=50) carry zero weights, so they store relu(0)=0;
    // all 128 K positions are written -> no separate zero-fill needed.
    __bf16* hidBase = &sHid[wave * 16 * KP];
    #pragma unroll
    for (int nt = 0; nt < 4; nt++) {
        int col = nt * 16 + r;
        float bap = (col < HID) ? b_AP[col] : 0.f;
        float bpp = (col < HID) ? b_PP[col] : 0.f;
        #pragma unroll
        for (int vv = 0; vv < 8; vv++) {
            int row = vv + 8 * g;   // C layout: VGPR v -> M = v + 8*(lane/16)
            hidBase[row * KP + col]        = (__bf16)fmaxf(cA[nt][vv] + bap, 0.f);
            hidBase[row * KP + NPAD + col] = (__bf16)fmaxf(cP[nt][vv] + bpp, 0.f);
        }
    }

    // ---- GEMM 2: out = hidden(128) @ Wp ----  (same-wave DS ops are in-order)
    v8f o[4] = {};
    const __bf16* hidRow = &sHid[(wave * 16 + r) * KP];
    #pragma unroll
    for (int kc = 0; kc < 4; kc++) {
        v16bf a = load_afrag(hidRow, kc * 32, g);
        #pragma unroll
        for (int nt = 0; nt < 4; nt++) {
            v16bf bw = load_bfragT(&sW[WT_WP], KP, nt * 16 + r, kc * 32, g);
            o[nt] = __builtin_amdgcn_wmma_f32_16x16x32_bf16(
                false, a, false, bw, (short)0, o[nt], false, false);
        }
    }

    // bias + relu + pair_mask, store P (16 consecutive f32 per half-wave row)
    const float* pm  = pair_mask + (size_t)(b * M_SZ + i) * M_SZ + j0;
    float*       out = Pout + ((size_t)(b * M_SZ + i) * M_SZ + j0) * OUT_P;
    #pragma unroll
    for (int nt = 0; nt < 4; nt++) {
        int col = nt * 16 + r;
        if (col < OUT_P) {
            float bp = b_P[col];
            #pragma unroll
            for (int vv = 0; vv < 8; vv++) {
                int row = vv + 8 * g;
                out[row * OUT_P + col] = fmaxf(o[nt][vv] + bp, 0.f) * pm[row];
            }
        }
    }
}

// --------------------------------------------------------------------------
extern "C" void kernel_launch(void* const* d_in, const int* in_sizes, int n_in,
                              void* d_out, int out_size, void* d_ws, size_t ws_size,
                              hipStream_t stream)
{
    const float* atom   = (const float*)d_in[0];
    const float* pairf  = (const float*)d_in[1];
    const float* amask  = (const float*)d_in[2];
    const float* pmask  = (const float*)d_in[3];
    const float* W_AA   = (const float*)d_in[4];
    const float* b_AA   = (const float*)d_in[5];
    const float* W_PA   = (const float*)d_in[6];
    const float* b_PA   = (const float*)d_in[7];
    const float* W_A    = (const float*)d_in[8];
    const float* b_A    = (const float*)d_in[9];
    const float* W_AP   = (const float*)d_in[10];
    const float* b_AP   = (const float*)d_in[11];
    const float* W_PP   = (const float*)d_in[12];
    const float* b_PP   = (const float*)d_in[13];
    const float* W_P    = (const float*)d_in[14];
    const float* b_P    = (const float*)d_in[15];

    float* Aout = (float*)d_out;
    float* Pout = Aout + (size_t)B_SZ * M_SZ * OUT_A;
    __bf16* wpack = (__bf16*)d_ws;

    pack_weights<<<(WT_TOT + 255) / 256, 256, 0, stream>>>(W_AP, W_PP, W_P, wpack);
    atom_kernel<<<B_SZ * M_SZ, 64, 0, stream>>>(atom, pairf, amask,
                                                W_AA, b_AA, W_PA, b_PA, W_A, b_A, Aout);
    pair_kernel<<<B_SZ * M_SZ, 256, 0, stream>>>(atom, pairf, pmask, wpack,
                                                 b_AP, b_PP, b_P, Pout);
}
